// Network_63969242907108
// MI455X (gfx1250) — matrix operand order repaired and checked by reading, not compiled, not run on hardware
//
#include <hip/hip_runtime.h>

// ---------------------------------------------------------------------------
// SLAYER N-MNIST net forward on gfx1250 (wave32, WMMA 16x16x32 f16 -> f32).
//
// Pipeline (all intermediates time-major [B,T,C,H,W], spikes stored f16):
//   transpose_in : [B,2,34,34,T] f32 -> [B*T, 2,34,34] f16
//   pack_w       : weights f32 -> f16 A-tiles in WMMA lane layout (per layer)
//   conv_wmma<>  : implicit-GEMM conv per image, fully unrolled WMMA chain
//   spike_f16    : per-neuron refractory recurrence over T (u f32 -> s f16)
//   pool_scale   : 2x2 sum * 1.1*theta (s f16 -> u f32)
//   dense_wmma<> : [2400 x 4096] x [4096 x 16] via WMMA
//   spike_out    : final recurrence + transpose back to [B,10,T] f32
// ---------------------------------------------------------------------------

typedef __attribute__((ext_vector_type(16))) _Float16 v16h;
typedef __attribute__((ext_vector_type(8)))  float    v8f;

#define THETA 10.0f

// REF_KERNEL[1..10] for tau=1, ts=1, mult=-2*theta:  -20*t*exp(1-t)
__constant__ float c_refk[10] = {
    -20.0f,
    -14.715177646857693f,
    -8.1201169941967617f,
    -3.9829654694291155f,
    -1.8315638888734179f,
    -0.80855363989025604f,
    -0.34702530473329019f,
    -0.14590111448872260f,
    -0.060383273022452133f,
    -0.024681960817335920f};

// --------------------------------------------------------------------------
// Input transpose: [B, CHW, T] f32 (events 0/1) -> [B*T, CHW] f16 (exact).
__global__ __launch_bounds__(256) void transpose_in(
    const float* __restrict__ in, _Float16* __restrict__ X, int CHW, int T) {
  int p = blockIdx.x;              // p = b*CHW + q
  int b = p / CHW;
  int q = p - b * CHW;
  for (int t = threadIdx.x; t < T; t += blockDim.x) {
    float v = in[(size_t)p * T + t];
    X[((size_t)b * T + t) * CHW + q] = (_Float16)v;
  }
}

// --------------------------------------------------------------------------
// Pack weights [O, Kreal] f32 into f16 WMMA A-tiles.
// Tile (mb, kt) = 512 halfs; element (lane, i):
//   M = mb*16 + lane%16,  K = kt*32 + i + 8*((i>=8) + (lane>=16))
// matching the 16-bit 16x32 A-matrix VGPR layout (ISA 7.12.2).
__global__ __launch_bounds__(256) void pack_w(
    const float* __restrict__ w, _Float16* __restrict__ wp,
    int O, int Kreal, int nKt, int total) {
  int idx = blockIdx.x * blockDim.x + threadIdx.x;
  if (idx >= total) return;
  int e  = idx & 511;
  int t_ = idx >> 9;
  int kt = t_ % nKt;
  int mb = t_ / nKt;
  int lane = e >> 4;
  int i    = e & 15;
  int M = mb * 16 + (lane & 15);
  int K = kt * 32 + i + 8 * ((i >> 3) + (lane >> 4));
  float v = 0.0f;
  if (M < O && K < Kreal) v = w[(size_t)M * Kreal + K];
  wp[idx] = (_Float16)v;
}

// --------------------------------------------------------------------------
// Implicit-GEMM conv, one wave per (image, 16-pixel N tile, 16-row M block).
// All geometry is compile-time: divisions fold to shifts, k-loop fully
// unrolls into a back-to-back v_wmma_f32_16x16x32_f16 chain.
// A = packed weights (f16); B = spike patch built in registers:
//   lane%16 = output pixel n, half i = K row (c,kh,kw) in A's table order.
template <int Cin, int Hin, int Win, int O, int Hout, int Wout,
          int KSZ, int PAD, int Kreal, int nKt, int nNt, int nMb>
__global__ __launch_bounds__(256) void conv_wmma(
    const _Float16* __restrict__ X, const _Float16* __restrict__ Wp,
    float* __restrict__ U, int NIMG) {
  int gid  = blockIdx.x * blockDim.x + threadIdx.x;
  int wave = gid >> 5;
  int lane = threadIdx.x & 31;
  if (wave >= NIMG * nNt * nMb) return;
  int mb  = wave % nMb;        // nMb is a power of two here -> mask
  int t1  = wave / nMb;
  int nt  = t1 % nNt;          // nNt is a power of two here -> mask
  int img = t1 / nNt;

  // Output-pixel coordinates for this wave's 16-column N tile (pad folded in).
  int n0  = nt * 16 + (lane & 15);
  int oh0 = n0 / Wout - PAD;
  int ow0 = n0 % Wout - PAD;

  const _Float16* xim = X + (size_t)img * (Cin * Hin * Win);
  const _Float16* wp  = Wp + (size_t)mb * (nKt * 512);
  v8f acc = {};
#pragma unroll
  for (int kt = 0; kt < nKt; ++kt) {
    v16h a = *(const v16h*)(wp + kt * 512 + lane * 16);
    v16h b = {};
#pragma unroll
    for (int i = 0; i < 16; ++i) {
      int k = kt * 32 + i + 8 * ((i >> 3) + (lane >> 4));
      if (k < Kreal) {
        int c   = k / (KSZ * KSZ);
        int rem = k - c * (KSZ * KSZ);
        int kh  = rem / KSZ;
        int kw  = rem - kh * KSZ;
        const _Float16* xc = xim + c * (Hin * Win);
        int ih = oh0 + kh;
        int iw = ow0 + kw;
        _Float16 v = (_Float16)0.0f;
        if (ih >= 0 && ih < Hin && iw >= 0 && iw < Win) v = xc[ih * Win + iw];
        b[i] = v;
      }
    }
    acc = __builtin_amdgcn_wmma_f32_16x16x32_f16(
        false, a, false, b, (short)0, acc, false, false);
  }
  // D layout: VGPR r -> (M = r + 8*(lane/16), N = lane%16)
  int n     = nt * 16 + (lane & 15);
  int mbase = mb * 16 + ((lane >> 4) << 3);
  float* uim = U + (size_t)img * (O * Hout * Wout);
#pragma unroll
  for (int r = 0; r < 8; ++r) {
    uim[(mbase + r) * (Hout * Wout) + n] = acc[r];
  }
}

// --------------------------------------------------------------------------
// Dense: u[b,t,o] = sum_k wfc[o,k] * s5[b,t,k];  M=16 (10 padded), K=4096,
// N = image index (2400 = 150 tiles of 16). Writes padded [NIMG,16] f32.
template <int CHW, int nKt>
__global__ __launch_bounds__(256) void dense_wmma(
    const _Float16* __restrict__ S, const _Float16* __restrict__ Wp,
    float* __restrict__ U, int nNt) {
  int gid  = blockIdx.x * blockDim.x + threadIdx.x;
  int wave = gid >> 5;
  int lane = threadIdx.x & 31;
  if (wave >= nNt) return;
  int nt = wave;

  v8f acc = {};
#pragma unroll 4
  for (int kt = 0; kt < nKt; ++kt) {
    v16h a = *(const v16h*)(Wp + kt * 512 + lane * 16);
    v16h b;
    int n = nt * 16 + (lane & 15);
#pragma unroll
    for (int i = 0; i < 16; ++i) {
      int k = kt * 32 + i + 8 * ((i >> 3) + (lane >> 4));
      b[i] = S[(size_t)n * CHW + k];
    }
    acc = __builtin_amdgcn_wmma_f32_16x16x32_f16(
        false, a, false, b, (short)0, acc, false, false);
  }
  int n     = nt * 16 + (lane & 15);
  int mbase = (lane >> 4) << 3;
#pragma unroll
  for (int r = 0; r < 8; ++r) {
    U[(size_t)n * 16 + mbase + r] = acc[r];
  }
}

// --------------------------------------------------------------------------
// Spike recurrence: one thread per neuron, sequential over T with a 10-deep
// pending-refractory shift register.  u f32 [B*T, N] -> s f16 [B*T, N].
__global__ __launch_bounds__(256) void spike_f16(
    const float* __restrict__ U, _Float16* __restrict__ S,
    int B_, int N, int T) {
  int id = blockIdx.x * blockDim.x + threadIdx.x;
  if (id >= B_ * N) return;
  int b  = id / N;
  int nn = id - b * N;
  const float* u = U + (size_t)b * T * N + nn;
  _Float16*    s = S + (size_t)b * T * N + nn;
  float buf[10];
#pragma unroll
  for (int i = 0; i < 10; ++i) buf[i] = 0.0f;
  for (int t = 0; t < T; ++t) {
    float um = u[(size_t)t * N] + buf[0];
    float ev = (um >= THETA) ? 1.0f : 0.0f;
#pragma unroll
    for (int i = 0; i < 9; ++i) buf[i] = buf[i + 1] + ev * c_refk[i];
    buf[9] = ev * c_refk[9];
    s[(size_t)t * N] = (_Float16)ev;  // amplitude 1/Ts, Ts=1
  }
}

// --------------------------------------------------------------------------
// 2x2 sum pool * 1.1*theta:  s f16 [NIMG,C,H,W] -> u f32 [NIMG,C,H/2,W/2].
__global__ __launch_bounds__(256) void pool_scale(
    const _Float16* __restrict__ S, float* __restrict__ U,
    int NIMG, int C, int H, int W) {
  int Ho = H >> 1, Wo = W >> 1;
  size_t total = (size_t)NIMG * C * Ho * Wo;
  size_t idx = (size_t)blockIdx.x * blockDim.x + threadIdx.x;
  if (idx >= total) return;
  int ow  = (int)(idx % Wo);
  int oh  = (int)((idx / Wo) % Ho);
  int c   = (int)((idx / ((size_t)Wo * Ho)) % C);
  int img = (int)(idx / ((size_t)Wo * Ho * C));
  const _Float16* s = S + ((size_t)img * C + c) * H * W;
  int r0 = (2 * oh) * W + 2 * ow;
  float sum = (float)s[r0] + (float)s[r0 + 1] +
              (float)s[r0 + W] + (float)s[r0 + W + 1];
  U[idx] = 1.1f * THETA * sum;
}

// --------------------------------------------------------------------------
// Final spike + layout restore: u f32 [B*T, 16] -> out f32 [B, 10, T].
__global__ __launch_bounds__(128) void spike_out(
    const float* __restrict__ U, float* __restrict__ out, int B_, int T) {
  int id = blockIdx.x * blockDim.x + threadIdx.x;
  if (id >= B_ * 10) return;
  int b = id / 10;
  int o = id - b * 10;
  float buf[10];
#pragma unroll
  for (int i = 0; i < 10; ++i) buf[i] = 0.0f;
  for (int t = 0; t < T; ++t) {
    float um = U[((size_t)b * T + t) * 16 + o] + buf[0];
    float ev = (um >= THETA) ? 1.0f : 0.0f;
#pragma unroll
    for (int i = 0; i < 9; ++i) buf[i] = buf[i + 1] + ev * c_refk[i];
    buf[9] = ev * c_refk[9];
    out[(size_t)id * T + t] = ev;
  }
}

// ---------------------------------------------------------------------------
extern "C" void kernel_launch(void* const* d_in, const int* in_sizes, int n_in,
                              void* d_out, int out_size, void* d_ws, size_t ws_size,
                              hipStream_t stream) {
  (void)in_sizes; (void)n_in; (void)out_size; (void)ws_size;
  const float* spikeIn = (const float*)d_in[0];  // [8,2,34,34,300]
  const float* w1  = (const float*)d_in[1];      // [16,2,5,5]
  const float* w2  = (const float*)d_in[2];      // [32,16,3,3]
  const float* w3  = (const float*)d_in[3];      // [64,32,3,3]
  const float* wfc = (const float*)d_in[4];      // [10,64,8,8]
  float* out = (float*)d_out;                    // [8,10,1,1,300]

  const int B = 8, T = 300, NIMG = B * T;

  // Carve workspace (256B-aligned regions).
  char* ws = (char*)d_ws;
  size_t off = 0;
  auto carve = [&](size_t bytes) -> void* {
    void* p = ws + off;
    off = (off + bytes + 255) & ~(size_t)255;
    return p;
  };
  _Float16* X    = (_Float16*)carve((size_t)NIMG * 2312 * 2);   // input f16
  _Float16* Wp1  = (_Float16*)carve((size_t)1 * 2 * 512 * 2);
  _Float16* Wp2  = (_Float16*)carve((size_t)2 * 5 * 512 * 2);
  _Float16* Wp3  = (_Float16*)carve((size_t)4 * 9 * 512 * 2);
  _Float16* Wpfc = (_Float16*)carve((size_t)1 * 128 * 512 * 2);
  float*    Ua   = (float*)carve((size_t)NIMG * 16384 * 4);     // 157 MB
  float*    Ub   = (float*)carve((size_t)NIMG * 4096 * 4);      //  39 MB
  _Float16* Sa   = (_Float16*)carve((size_t)NIMG * 16384 * 2);  //  79 MB
  _Float16* Sb   = (_Float16*)carve((size_t)NIMG * 4096 * 2);   //  20 MB

  // 0) transpose input to time-major f16
  transpose_in<<<B * 2312, 256, 0, stream>>>(spikeIn, X, 2312, T);

  // 1) pack weights into WMMA A-tiles
  pack_w<<<(1 * 2 * 512) / 256, 256, 0, stream>>>(w1, Wp1, 16, 50, 2, 1 * 2 * 512);
  pack_w<<<(2 * 5 * 512) / 256, 256, 0, stream>>>(w2, Wp2, 32, 144, 5, 2 * 5 * 512);
  pack_w<<<(4 * 9 * 512) / 256, 256, 0, stream>>>(w3, Wp3, 64, 288, 9, 4 * 9 * 512);
  pack_w<<<(128 * 512) / 256, 256, 0, stream>>>(wfc, Wpfc, 10, 4096, 128, 128 * 512);

  // 2) conv1 (2->16, 34x34 -> 32x32, 5x5 pad1): waves = 2400*64*1
  conv_wmma<2, 34, 34, 16, 32, 32, 5, 1, 50, 2, 64, 1>
      <<<(NIMG * 64 * 1 + 7) / 8, 256, 0, stream>>>(X, Wp1, Ua, NIMG);
  spike_f16<<<(B * 16384 + 255) / 256, 256, 0, stream>>>(Ua, Sa, B, 16384, T);

  // 3) pool -> spike: [16,32,32] -> [16,16,16]
  pool_scale<<<(int)(((size_t)NIMG * 16 * 16 * 16 + 255) / 256), 256, 0, stream>>>(
      Sa, Ub, NIMG, 16, 32, 32);
  spike_f16<<<(B * 4096 + 255) / 256, 256, 0, stream>>>(Ub, Sb, B, 4096, T);

  // 4) conv2 (16->32, 16x16, 3x3 pad1): waves = 2400*16*2
  conv_wmma<16, 16, 16, 32, 16, 16, 3, 1, 144, 5, 16, 2>
      <<<(NIMG * 16 * 2 + 7) / 8, 256, 0, stream>>>(Sb, Wp2, Ua, NIMG);
  spike_f16<<<(B * 8192 + 255) / 256, 256, 0, stream>>>(Ua, Sa, B, 8192, T);

  // 5) pool -> spike: [32,16,16] -> [32,8,8]
  pool_scale<<<(int)(((size_t)NIMG * 32 * 8 * 8 + 255) / 256), 256, 0, stream>>>(
      Sa, Ub, NIMG, 32, 16, 16);
  spike_f16<<<(B * 2048 + 255) / 256, 256, 0, stream>>>(Ub, Sb, B, 2048, T);

  // 6) conv3 (32->64, 8x8, 3x3 pad1): waves = 2400*4*4
  conv_wmma<32, 8, 8, 64, 8, 8, 3, 1, 288, 9, 4, 4>
      <<<(NIMG * 4 * 4 + 7) / 8, 256, 0, stream>>>(Sb, Wp3, Ua, NIMG);
  spike_f16<<<(B * 4096 + 255) / 256, 256, 0, stream>>>(Ua, Sa, B, 4096, T);

  // 7) dense (4096 -> 10, N = 2400 images): 150 waves
  dense_wmma<4096, 128><<<(150 + 7) / 8, 256, 0, stream>>>(Sa, Wpfc, Ub, 150);

  // 8) final spike + transpose back to [B,10,T]
  spike_out<<<1, 128, 0, stream>>>(Ub, out, B, T);
}
